// MultiHeadSelfAttention_4698694221952
// MI455X (gfx1250) — matrix operand order repaired and checked by reading, not compiled, not run on hardware
//
#include <hip/hip_runtime.h>
#include <hip/hip_bf16.h>

typedef __attribute__((ext_vector_type(16))) __bf16 v16bf;
typedef __attribute__((ext_vector_type(8)))  __bf16 v8bf;
typedef __attribute__((ext_vector_type(8)))  float  v8f;
typedef __attribute__((ext_vector_type(8)))  float  v8fx;   // 32B-aligned fp32 chunk

#define B_  4
#define S_  2048
#define D_  1024
#define E_  1024
#define H_  16
#define HD_ 64
#define PSTRIDE (S_ + 16)   // padded probs row stride: row r starts on bank (16r)%64

// A-fragment K index for 16-bit 16x32 layout (wave32):
// lanes 0-15: e<8 -> K=e, e>=8 -> K=e+8 ; lanes 16-31: +8
__device__ __forceinline__ int a_kidx(int e, int hi) {
    return e + ((e >= 8) ? 8 : 0) + hi * 8;
}

// ---------------------------------------------------------------------------
// Kernel 1: qkv = x @ qkv_w.T + qkv_b.
//   Q (pre-scaled by 1/sqrt(HD)) -> bf16 [B,H,S,HD]
//   K                            -> bf16 [B,H,S,HD]
//   V                            -> bf16 [B,H,HD,S]  (transposed)
// ---------------------------------------------------------------------------
__global__ void qkv_kernel(const float* __restrict__ x,
                           const float* __restrict__ qkv_w,
                           const float* __restrict__ qkv_b,
                           __bf16* __restrict__ q_ws,
                           __bf16* __restrict__ k_ws,
                           __bf16* __restrict__ v_wsT) {
    const int wave = threadIdx.x >> 5;
    const int lane = threadIdx.x & 31;
    const int hi   = lane >> 4;
    const int ln   = lane & 15;
    const int m0 = blockIdx.x * 16;
    const int n0 = (blockIdx.y * 4 + wave) * 16;
    const int n  = n0 + ln;

    v8f acc = {};
    const float* xrow = x + (size_t)(m0 + ln) * D_;
    const float* wrow = qkv_w + (size_t)n * D_;

    for (int k0 = 0; k0 < D_; k0 += 32) {
        if (k0 + 32 < D_) {   // prefetch next k-step (global_prefetch_b8)
            __builtin_prefetch(xrow + k0 + 32 + hi * 8, 0, 1);
            __builtin_prefetch(wrow + k0 + 32 + hi * 16, 0, 1);
        }
        v16bf a;
        {
            v8fx p0 = *(const v8fx*)(xrow + k0 + hi * 16);       // e = 0..7
            v8fx p1 = *(const v8fx*)(xrow + k0 + hi * 16 + 8);   // e = 8..15
            #pragma unroll
            for (int e = 0; e < 8; ++e) { a[e] = (__bf16)p0[e]; a[8 + e] = (__bf16)p1[e]; }
        }
        v16bf bfrag;
        {
            v8fx p0 = *(const v8fx*)(wrow + k0 + hi * 16);
            v8fx p1 = *(const v8fx*)(wrow + k0 + hi * 16 + 8);
            #pragma unroll
            for (int e = 0; e < 8; ++e) { bfrag[e] = (__bf16)p0[e]; bfrag[8 + e] = (__bf16)p1[e]; }
        }
        acc = __builtin_amdgcn_wmma_f32_16x16x32_bf16(
                  false, a, false, bfrag, (short)0, acc, false, false);
    }

    const float bias = qkv_b[n];
    const int head = n / 192;
    const int j192 = n % 192;
    const int kind = j192 / 64;       // 0=q 1=k 2=v
    const int d    = j192 % 64;
    const int bb = m0 / S_;                   // tile never crosses batch boundary
    const int s0 = (m0 % S_) + hi * 8;

    if (kind == 2) {
        v8bf pack;                            // transposed V: one 16B store
        #pragma unroll
        for (int j = 0; j < 8; ++j) pack[j] = (__bf16)(acc[j] + bias);
        *(v8bf*)(v_wsT + ((size_t)(bb * H_ + head) * HD_ + d) * S_ + s0) = pack;
    } else {
        __bf16* dst = (kind == 0) ? q_ws : k_ws;
        const float scale = (kind == 0) ? 0.125f : 1.0f;
        #pragma unroll
        for (int j = 0; j < 8; ++j)
            dst[((size_t)(bb * H_ + head) * S_ + s0 + j) * HD_ + d] =
                (__bf16)((acc[j] + bias) * scale);
    }
}

// ---------------------------------------------------------------------------
// Kernel 2: fused attention. One block (4 waves) per (b, 16-row q tile),
// loops over all 16 heads so attention_mean needs no atomics.
// LDS: probs[16][PSTRIDE] fp32 (~129KB, bank-skewed) + scratch.
// ---------------------------------------------------------------------------
__global__ void attn_kernel(const float* __restrict__ matrix_mask,
                            const __bf16* __restrict__ q_ws,
                            const __bf16* __restrict__ k_ws,
                            const __bf16* __restrict__ v_wsT,
                            __bf16* __restrict__ values_ws,
                            float* __restrict__ attn_mean) {
    extern __shared__ float smem[];
    float* probs = smem;                       // 16 * PSTRIDE (padded)
    float* red   = smem + 16 * PSTRIDE;        // 16 * 8
    float* vacc  = red + 16 * 8;               // 4 * 16 * 64

    const int wave = threadIdx.x >> 5;
    const int lane = threadIdx.x & 31;
    const int hi   = lane >> 4;
    const int ln   = lane & 15;
    const int b  = blockIdx.x / (S_ / 16);
    const int q0 = (blockIdx.x % (S_ / 16)) * 16;
    const int c0 = wave * 512;                 // this wave's 512-column slice

    for (int h = 0; h < H_; ++h) {
        const size_t head_base = ((size_t)(b * H_ + h)) * S_ * HD_;

        // ---- Q A-fragments (HD=64 -> two k-steps of 32)
        v16bf qa[2];
        {
            const __bf16* qrow = q_ws + head_base + (size_t)(q0 + ln) * HD_;
            #pragma unroll
            for (int ks = 0; ks < 2; ++ks) {
                v8bf p0 = *(const v8bf*)(qrow + ks * 32 + hi * 8);
                v8bf p1 = *(const v8bf*)(qrow + ks * 32 + 16 + hi * 8);
                #pragma unroll
                for (int e = 0; e < 8; ++e) { qa[ks][e] = p0[e]; qa[ks][8 + e] = p1[e]; }
            }
        }

        // ---- stage 1: logits = Q K^T + mask (16 x 2048, wave covers 512)
        for (int ct = 0; ct < 32; ++ct) {
            const int col = c0 + ct * 16;
            const __bf16* krow = k_ws + head_base + (size_t)(col + ln) * HD_;
            if (ct + 1 < 32)
                __builtin_prefetch(krow + 16 * HD_, 0, 1);   // next K tile row
            v8f acc = {};
            #pragma unroll
            for (int ks = 0; ks < 2; ++ks) {
                v16bf kb;
                {
                    v8bf k0v = *(const v8bf*)(krow + ks * 32 + hi * 8);
                    v8bf k1v = *(const v8bf*)(krow + ks * 32 + 16 + hi * 8);
                    #pragma unroll
                    for (int e = 0; e < 8; ++e) { kb[e] = k0v[e]; kb[8 + e] = k1v[e]; }
                }
                acc = __builtin_amdgcn_wmma_f32_16x16x32_bf16(
                          false, qa[ks], false, kb, (short)0, acc, false, false);
            }
            const int cN = col + ln;
            const float* mcol = matrix_mask + ((size_t)b * S_ + q0 + hi * 8) * S_ + cN;
            #pragma unroll
            for (int j = 0; j < 8; ++j) {
                const int r = j + hi * 8;
                probs[r * PSTRIDE + cN] = acc[j] + mcol[(size_t)j * S_];
            }
        }
        __syncthreads();

        // ---- stage 2: row softmax (8 threads/row, stride-8 interleave ->
        //      32 lanes of a wave hit 32 distinct LDS banks) + attention_mean
        {
            const int r   = threadIdx.x >> 3;
            const int sub = threadIdx.x & 7;
            float* row = probs + r * PSTRIDE;
            float mx = -3.4e38f;
            for (int i = 0; i < 256; ++i) mx = fmaxf(mx, row[sub + 8 * i]);
            red[r * 8 + sub] = mx;
            __syncthreads();
            float rmax = red[r * 8];
            #pragma unroll
            for (int i = 1; i < 8; ++i) rmax = fmaxf(rmax, red[r * 8 + i]);
            float sum = 0.f;
            for (int i = 0; i < 256; ++i) {
                const int c = sub + 8 * i;
                const float e = __expf(row[c] - rmax);
                row[c] = e;
                sum += e;
            }
            __syncthreads();                   // rmax reads done before red reuse
            red[r * 8 + sub] = sum;
            __syncthreads();
            float rsum = 0.f;
            #pragma unroll
            for (int i = 0; i < 8; ++i) rsum += red[r * 8 + i];
            const float inv = 1.0f / rsum;
            float* am = attn_mean + ((size_t)b * S_ + (q0 + r)) * S_;
            for (int i = 0; i < 256; ++i) {
                const int c = sub + 8 * i;
                const float p = row[c] * inv;
                row[c] = p;
                const float contrib = p * (1.0f / H_);
                am[c] = (h == 0) ? contrib : (am[c] + contrib);  // block-private, L2-resident
            }
        }
        __syncthreads();

        // ---- stage 3: P @ V over this wave's 512-col K slice -> 16x64 partial
        const __bf16* vtb = v_wsT + head_base;  // [HD][S] for this (b,h)
        v8f oacc[4] = {};
        for (int kt = 0; kt < 16; ++kt) {
            const int k0 = c0 + kt * 32;
            if (kt + 1 < 16)
                __builtin_prefetch(vtb + (size_t)ln * S_ + k0 + 32, 0, 1);
            v16bf pa;
            {
                const float* prow = probs + ln * PSTRIDE;
                #pragma unroll
                for (int e = 0; e < 16; ++e)
                    pa[e] = (__bf16)prow[k0 + hi * 16 + e];
            }
            #pragma unroll
            for (int nt = 0; nt < 4; ++nt) {
                const __bf16* vcol = vtb + (size_t)(nt * 16 + ln) * S_;
                v16bf vb = *(const v16bf*)(vcol + k0 + hi * 16);  // contiguous along S
                oacc[nt] = __builtin_amdgcn_wmma_f32_16x16x32_bf16(
                               false, pa, false, vb, (short)0, oacc[nt], false, false);
            }
        }
        // cross-wave reduction of the 16x64 tile via LDS
        #pragma unroll
        for (int nt = 0; nt < 4; ++nt) {
            #pragma unroll
            for (int j = 0; j < 8; ++j) {
                const int r = j + hi * 8;
                vacc[(wave * 16 + r) * 64 + nt * 16 + ln] = oacc[nt][j];
            }
        }
        __syncthreads();
        #pragma unroll
        for (int u = 0; u < 8; ++u) {
            const int idx = threadIdx.x * 8 + u;     // 0..1023 over 16x64
            const int r = idx >> 6;
            const int nn = idx & 63;
            float s = 0.f;
            #pragma unroll
            for (int w2 = 0; w2 < 4; ++w2) s += vacc[(w2 * 16 + r) * 64 + nn];
            values_ws[((size_t)(b * S_) + q0 + r) * E_ + h * HD_ + nn] = (__bf16)s;
        }
        __syncthreads();   // probs/vacc reused next head
    }
}

// ---------------------------------------------------------------------------
// Kernel 3: o = values @ o_w.T + o_b   (fp32 out)
// ---------------------------------------------------------------------------
__global__ void oproj_kernel(const __bf16* __restrict__ values_ws,
                             const float* __restrict__ o_w,
                             const float* __restrict__ o_b,
                             float* __restrict__ out) {
    const int wave = threadIdx.x >> 5;
    const int lane = threadIdx.x & 31;
    const int hi   = lane >> 4;
    const int ln   = lane & 15;
    const int m0 = blockIdx.x * 16;
    const int n0 = (blockIdx.y * 4 + wave) * 16;
    const int n  = n0 + ln;

    v8f acc = {};
    const __bf16* arow = values_ws + (size_t)(m0 + ln) * E_;
    const float*  wrow = o_w + (size_t)n * E_;

    for (int k0 = 0; k0 < E_; k0 += 32) {
        if (k0 + 32 < E_) {
            __builtin_prefetch(arow + k0 + 32 + hi * 8, 0, 1);
            __builtin_prefetch(wrow + k0 + 32 + hi * 16, 0, 1);
        }
        v16bf a;
        {
            v8bf lo = *(const v8bf*)(arow + k0 + hi * 16);
            v8bf hh = *(const v8bf*)(arow + k0 + hi * 16 + 8);
            #pragma unroll
            for (int e = 0; e < 8; ++e) { a[e] = lo[e]; a[8 + e] = hh[e]; }
        }
        v16bf bfrag;
        {
            v8fx p0 = *(const v8fx*)(wrow + k0 + hi * 16);
            v8fx p1 = *(const v8fx*)(wrow + k0 + hi * 16 + 8);
            #pragma unroll
            for (int e = 0; e < 8; ++e) { bfrag[e] = (__bf16)p0[e]; bfrag[8 + e] = (__bf16)p1[e]; }
        }
        acc = __builtin_amdgcn_wmma_f32_16x16x32_bf16(
                  false, a, false, bfrag, (short)0, acc, false, false);
    }

    const float bias = o_b[n];
    #pragma unroll
    for (int j = 0; j < 8; ++j) {
        const int m = m0 + j + hi * 8;
        out[(size_t)m * E_ + n] = acc[j] + bias;
    }
}

// ---------------------------------------------------------------------------
extern "C" void kernel_launch(void* const* d_in, const int* in_sizes, int n_in,
                              void* d_out, int out_size, void* d_ws, size_t ws_size,
                              hipStream_t stream) {
    const float* x           = (const float*)d_in[0];
    // d_in[1] = src_key_padding_mask (unused by the module)
    const float* matrix_mask = (const float*)d_in[2];
    const float* qkv_w       = (const float*)d_in[3];
    const float* qkv_b       = (const float*)d_in[4];
    const float* o_w         = (const float*)d_in[5];
    const float* o_b         = (const float*)d_in[6];

    float* o_out  = (float*)d_out;                       // [B,S,E]
    float* am_out = o_out + (size_t)B_ * S_ * E_;        // [B,S,S]

    const size_t per_tensor = (size_t)B_ * H_ * S_ * HD_;  // 8.4M bf16 elems each
    __bf16* q_ws      = (__bf16*)d_ws;
    __bf16* k_ws      = q_ws + per_tensor;
    __bf16* v_wsT     = k_ws + per_tensor;               // [B,H,HD,S]
    __bf16* values_ws = v_wsT + per_tensor;              // [B*S, E]

    dim3 blk(128);
    qkv_kernel<<<dim3((B_ * S_) / 16, (3 * E_) / 64), blk, 0, stream>>>(
        x, qkv_w, qkv_b, q_ws, k_ws, v_wsT);

    const size_t smem = (size_t)(16 * PSTRIDE + 16 * 8 + 4 * 16 * 64) * sizeof(float);
    attn_kernel<<<dim3(B_ * (S_ / 16)), blk, smem, stream>>>(
        matrix_mask, q_ws, k_ws, v_wsT, values_ws, am_out);

    oproj_kernel<<<dim3((B_ * S_) / 16, E_ / 64), blk, 0, stream>>>(
        values_ws, o_w, o_b, o_out);
}
